// CausalQKMemoryProjection_58798102282469
// MI455X (gfx1250) — matrix operand order repaired and checked by reading, not compiled, not run on hardware
//
#include <hip/hip_runtime.h>
#include <hip/hip_bf16.h>

// CausalQKMemoryProjection for MI455X (gfx1250), wave32 + WMMA bf16.
// B=4, H=16, S=2048, Dh=64 -> D=1024, CTX=128, gamma=0.95.

#define BB   4
#define HH   16
#define SS   2048
#define DHD  64
#define DD   1024
#define CTXW 128
#define GAMMA 0.95f
#define NORMP (DD * 0.01f)
#define EPSV  1e-6f

#define QSTR  1032  // Q tile row stride (halfwords): 516 dwords == 4 mod 64 -> conflict-free
#define KSTR  136   // phase-1 K row stride (halfwords)
#define KTSTR 168   // phase-2 K^T row stride (halfwords)
#define DWSTR 168   // weighted-dots row stride (halfwords)
#define DCH   128   // d-chunk size

// dynamic LDS partition (u16 units)
#define OFF_Q   0
#define OFF_BIG (16 * QSTR)                        // 16512
#define OFF_DW  (OFF_BIG + 128 * KTSTR)            // 38016
#define OFF_F32 (OFF_DW + 16 * DWSTR)              // 40704 (even)
#define SMEM_BYTES (OFF_F32 * 2 + (160 + 160 + 128 + 16) * 4)   // 83264

typedef unsigned short u16;
typedef unsigned int   u32;
typedef __attribute__((ext_vector_type(16))) __bf16 v16bf;
typedef __attribute__((ext_vector_type(8)))  float  v8f;

struct U32x8 { uint4 a, b; };

__device__ __forceinline__ u16 f2bf(float f) {     // f32 -> bf16 RNE
  u32 u = __builtin_bit_cast(u32, f);
  u += 0x7FFFu + ((u >> 16) & 1u);
  return (u16)(u >> 16);
}

// B-matrix fragment (32x16 bf16): lane n=lane&15 holds column n; halfword h=k&15,
// k-half selected by lane-half -> one contiguous 32-byte read.
__device__ __forceinline__ v16bf frag_b(const u16* p) {
  U32x8 s;
  s.a = ((const uint4*)p)[0];
  s.b = ((const uint4*)p)[1];
  return __builtin_bit_cast(v16bf, s);
}

// A-matrix fragment (16x32 bf16): lane m=lane&15 holds row m; halfwords 0..7 at k0,
// halfwords 8..15 at k0+8 -> two 16-byte reads.
__device__ __forceinline__ v16bf frag_a(const u16* p0) {
  U32x8 s;
  s.a = *(const uint4*)p0;
  s.b = *(const uint4*)(p0 + 8);
  return __builtin_bit_cast(v16bf, s);
}

// Kernel 0: M (1024x1024 f32, row-major) -> MT bf16 (MT[n][d] = M[d][n]) in d_ws.
__global__ __launch_bounds__(256) void cqkmp_transpose(const float* __restrict__ M,
                                                       u16* __restrict__ MT) {
  __shared__ float tile[16][17];
  int tx = threadIdx.x & 15, ty = threadIdx.x >> 4;
  int bx = blockIdx.x, by = blockIdx.y;
  tile[ty][tx] = M[(size_t)(by * 16 + ty) * DD + bx * 16 + tx];
  __syncthreads();
  MT[(size_t)(bx * 16 + ty) * DD + by * 16 + tx] = f2bf(tile[tx][ty]);
}

__global__ __launch_bounds__(256) void cqkmp_main(const float* __restrict__ q,
                                                  const float* __restrict__ k,
                                                  const float* __restrict__ gates,
                                                  const u16*  __restrict__ MT,
                                                  float* __restrict__ out) {
  extern __shared__ __align__(16) u16 smem[];
  u16*   smQ    = smem + OFF_Q;      // 16 x QSTR  bf16 Q tile
  u16*   smBig  = smem + OFF_BIG;    // phase1: 144 x KSTR K rows; phase2: 128 x KTSTR K^T
  u16*   smDW   = smem + OFF_DW;     // 16 x DWSTR weighted dots
  float* smKn   = (float*)(smem + OFF_F32);
  float* smGate = smKn + 160;
  float* smGpow = smGate + 160;
  float* smNorm = smGpow + 128;

  const int tid  = threadIdx.x;
  const int lane = tid & 31;
  const int lm   = lane & 15;
  const int hi   = lane >> 4;
  const int wvu  = __builtin_amdgcn_readfirstlane(tid >> 5);  // scalar wave id 0..7

  const int wg = blockIdx.x;
  const int b  = wg >> 7;            // wg / (S/16)
  const int tq = wg & 127;
  const int T0 = tq << 4;
  const int jbase = T0 - CTXW;       // first key row of the band window

  const size_t SD = (size_t)SS * DHD;              // S * 64
  const float* qb = q + (size_t)b * HH * SD;
  const float* kb = k + (size_t)b * HH * SD;

  // ---------------- init: tables + stage Q tile (bf16) ----------------
  for (int i = tid; i < 16 * DWSTR; i += 256) smDW[i] = 0;
  for (int i = tid; i < 160; i += 256) {
    int j = jbase + i;
    smGate[i] = (j >= 0 && j < SS) ? gates[(size_t)b * SS + j] : 0.f;
    smKn[i]   = 0.f;
  }
  if (tid < CTXW) smGpow[tid] = __powf(GAMMA, (float)tid);
  if (tid < 16)   smNorm[tid] = NORMP;
  for (int idx = tid; idx < 16 * (DD / 4); idx += 256) {   // 16 rows x 256 float4
    int row = idx >> 8;
    int d   = (idx & 255) * 4;
    float4 v = *(const float4*)(qb + (size_t)(d >> 6) * SD + (size_t)(T0 + row) * DHD + (d & 63));
    u16* dst = &smQ[row * QSTR + d];
    dst[0] = f2bf(v.x); dst[1] = f2bf(v.y); dst[2] = f2bf(v.z); dst[3] = f2bf(v.w);
  }

  // ---------------- phase 1: banded dots = Q (16xD) . K^T ----------------
  v8f acc0 = {};     // tile wvu   (j rows [wvu*16, wvu*16+16) of band)
  v8f acc1 = {};     // tile 8     (wave 0 only)
  const u16* qfr = &smQ[lm * QSTR];
  for (int d0 = 0; d0 < DD; d0 += DCH) {
    __syncthreads();
    // stage K band rows [jbase, jbase+144) dims [d0, d0+128) as bf16, accumulate ||k||^2
    for (int idx = tid; idx < 144 * (DCH / 4); idx += 256) {
      int jj  = idx >> 5;
      int seg = idx & 31;
      int d   = d0 + seg * 4;
      int j   = jbase + jj;
      float4 v = make_float4(0.f, 0.f, 0.f, 0.f);
      if (j >= 0)
        v = *(const float4*)(kb + (size_t)(d >> 6) * SD + (size_t)j * DHD + (d & 63));
      u16* dst = &smBig[jj * KSTR + seg * 4];
      dst[0] = f2bf(v.x); dst[1] = f2bf(v.y); dst[2] = f2bf(v.z); dst[3] = f2bf(v.w);
      atomicAdd(&smKn[jj], v.x * v.x + v.y * v.y + v.z * v.z + v.w * v.w);
    }
    __syncthreads();
#pragma unroll
    for (int kk = 0; kk < DCH / 32; ++kk) {
      v16bf a = frag_a(qfr + d0 + kk * 32 + hi * 16);
      v16bf b0 = frag_b(&smBig[(wvu * 16 + lm) * KSTR + kk * 32 + hi * 16]);
      acc0 = __builtin_amdgcn_wmma_f32_16x16x32_bf16(false, a, false, b0, (short)0, acc0, false, false);
      if (wvu == 0) {   // scalar branch: EXEC stays all-ones for the WMMA
        v16bf b1 = frag_b(&smBig[(128 + lm) * KSTR + kk * 32 + hi * 16]);
        acc1 = __builtin_amdgcn_wmma_f32_16x16x32_bf16(false, a, false, b1, (short)0, acc1, false, false);
      }
    }
  }

  // ---------------- weights: dots *= gamma^(t-j-1)*gate[j]; norm_sum ----------------
  {
    int jl = wvu * 16 + lm;
    float gk = smGate[jl];
    float kn = smKn[jl];
    int j = jbase + jl;
#pragma unroll
    for (int r = 0; r < 8; ++r) {
      int m = r + hi * 8;
      int dist = (T0 + m) - j;
      float wgt = (dist >= 1 && dist <= CTXW) ? smGpow[dist - 1] * gk : 0.f;
      smDW[m * DWSTR + jl] = f2bf(acc0[r] * wgt);
      float contrib = wgt * kn;        // reduce over j within 16-lane group
      contrib += __shfl_xor(contrib, 1);
      contrib += __shfl_xor(contrib, 2);
      contrib += __shfl_xor(contrib, 4);
      contrib += __shfl_xor(contrib, 8);
      if (lm == 0) atomicAdd(&smNorm[m], contrib);
    }
  }
  if (wvu == 0) {      // tile 8 (diagonal block)
    int jl = 128 + lm;
    float gk = smGate[jl];
    float kn = smKn[jl];
    int j = jbase + jl;
#pragma unroll
    for (int r = 0; r < 8; ++r) {
      int m = r + hi * 8;
      int dist = (T0 + m) - j;
      float wgt = (dist >= 1 && dist <= CTXW) ? smGpow[dist - 1] * gk : 0.f;
      smDW[m * DWSTR + jl] = f2bf(acc1[r] * wgt);
      float contrib = wgt * kn;
      contrib += __shfl_xor(contrib, 1);
      contrib += __shfl_xor(contrib, 2);
      contrib += __shfl_xor(contrib, 4);
      contrib += __shfl_xor(contrib, 8);
      if (lm == 0) atomicAdd(&smNorm[m], contrib);
    }
  }
  __syncthreads();

  // zero K^T pad region (jj in [144, KTSTR)) once
  for (int idx = tid; idx < 128 * (KTSTR - 144); idx += 256) {
    int row = idx / (KTSTR - 144);
    int jj  = 144 + idx % (KTSTR - 144);
    smBig[row * KTSTR + jj] = 0;
  }

  // ---------------- phase 2: out = (q@M + dW@K) / (norm+eps) ----------------
  for (int n0 = 0; n0 < DD; n0 += 128) {
    __syncthreads();
    // stage K^T chunk: KT[d-n0][jj] = K[jbase+jj][d], bf16
    for (int idx = tid; idx < 144 * 32; idx += 256) {
      int jj  = idx >> 5;
      int seg = idx & 31;
      int d   = n0 + seg * 4;
      int j   = jbase + jj;
      float4 v = make_float4(0.f, 0.f, 0.f, 0.f);
      if (j >= 0)
        v = *(const float4*)(kb + (size_t)(d >> 6) * SD + (size_t)j * DHD + (d & 63));
      int r0 = seg * 4;
      smBig[(r0 + 0) * KTSTR + jj] = f2bf(v.x);
      smBig[(r0 + 1) * KTSTR + jj] = f2bf(v.y);
      smBig[(r0 + 2) * KTSTR + jj] = f2bf(v.z);
      smBig[(r0 + 3) * KTSTR + jj] = f2bf(v.w);
    }
    __syncthreads();

    const int ncol = n0 + wvu * 16;    // this wave's 16 output columns
    v8f c = {};

    // persistent term: Q(16x1024) @ M via MT rows (B column n = MT row ncol+n).
    // Software-pipelined: fragments for step kk+1 are loaded before the WMMA of kk.
    const u16* mrow = MT + (size_t)(ncol + lm) * DD;
    if (n0 + 128 < DD) {   // prefetch next chunk's MT row into caches
      __builtin_prefetch(MT + (size_t)(n0 + 128 + wvu * 16 + lm) * DD, 0, 0);
    }
    {
      v16bf a  = frag_a(qfr + hi * 16);
      v16bf bm = frag_b(mrow + hi * 16);
#pragma unroll 4
      for (int kk = 0; kk < 32; ++kk) {
        int kn_ = (kk + 1) & 31;   // wrapped: last iteration reloads step 0 (discarded)
        v16bf an = frag_a(qfr + kn_ * 32 + hi * 16);
        v16bf bn = frag_b(mrow + kn_ * 32 + hi * 16);
        c = __builtin_amdgcn_wmma_f32_16x16x32_bf16(false, a, false, bm, (short)0, c, false, false);
        a = an; bm = bn;
      }
    }
    // context term: dW(16x160) @ K(160x16) from KT, same pipelining
    {
      v16bf a  = frag_a(&smDW[lm * DWSTR + hi * 16]);
      v16bf bf = frag_b(&smBig[(wvu * 16 + lm) * KTSTR + hi * 16]);
#pragma unroll
      for (int kk = 0; kk < 5; ++kk) {
        int kn_ = (kk + 1) % 5;
        v16bf an = frag_a(&smDW[lm * DWSTR + kn_ * 32 + hi * 16]);
        v16bf bn = frag_b(&smBig[(wvu * 16 + lm) * KTSTR + kn_ * 32 + hi * 16]);
        c = __builtin_amdgcn_wmma_f32_16x16x32_bf16(false, a, false, bf, (short)0, c, false, false);
        a = an; bf = bn;
      }
    }

    // finalize: divide by norm, scatter to (B,H,S,Dh)
    int d = ncol + lm;
    float* orow = out + (size_t)b * HH * SD + (size_t)(d >> 6) * SD + (d & 63);
#pragma unroll
    for (int r = 0; r < 8; ++r) {
      int m = r + hi * 8;
      float inv = 1.f / (smNorm[m] + EPSV);
      orow[(size_t)(T0 + m) * DHD] = c[r] * inv;
    }
  }
}

extern "C" void kernel_launch(void* const* d_in, const int* in_sizes, int n_in,
                              void* d_out, int out_size, void* d_ws, size_t ws_size,
                              hipStream_t stream) {
  (void)in_sizes; (void)n_in; (void)out_size; (void)ws_size;
  const float* q     = (const float*)d_in[0];
  const float* k     = (const float*)d_in[1];
  const float* gates = (const float*)d_in[2];
  const float* M     = (const float*)d_in[3];
  u16* MT = (u16*)d_ws;   // 1024*1024 bf16 = 2 MB scratch

  cqkmp_transpose<<<dim3(DD / 16, DD / 16), 256, 0, stream>>>(M, MT);
  cqkmp_main<<<dim3(BB * (SS / 16)), 256, SMEM_BYTES, stream>>>(q, k, gates, MT, (float*)d_out);
}
